// SetCoverHolo_32770600469044
// MI455X (gfx1250) — matrix operand order repaired and checked by reading, not compiled, not run on hardware
//
#include <hip/hip_runtime.h>

typedef __attribute__((ext_vector_type(16))) __bf16 v16bf;
typedef __attribute__((ext_vector_type(8)))  __bf16 v8bf;
typedef __attribute__((ext_vector_type(8)))  float  v8f;
typedef unsigned int v4u __attribute__((ext_vector_type(4)));
typedef int v8i_t __attribute__((ext_vector_type(8)));
typedef int v4i_t __attribute__((ext_vector_type(4)));

// ---------------------------------------------------------------------------
// WMMA helpers (CDNA5 wave32, 16x16x32 bf16 -> f32 accumulate)
// ---------------------------------------------------------------------------
__device__ __forceinline__ v8f wmma_bf16(v16bf a, v16bf b, v8f c) {
  return __builtin_amdgcn_wmma_f32_16x16x32_bf16(false, a, false, b, (short)0, c,
                                                 false, false);
}

// A fragment: 16-bit A-matrix 16xK layout (ISA 7.12.2).
__device__ __forceinline__ v16bf load_a_frag(const __bf16* tile, int lane, int kt,
                                             int stride) {
  const int m    = lane & 15;
  const int koff = (lane < 16) ? 0 : 8;
  v16bf a;
#pragma unroll
  for (int j = 0; j < 8; ++j) {
    const int kk = ((j < 4) ? (2 * j) : (16 + 2 * (j - 4))) + koff;
    const int k  = kt * 32 + kk;
    a[2 * j]     = tile[m * stride + k];
    a[2 * j + 1] = tile[m * stride + k + 1];
  }
  return a;
}

// B fragment for Y = X @ W^T with W row-major [64][K]: B(k,n) = W[n*K + k].
__device__ __forceinline__ v16bf load_b_frag(const __bf16* W, int lane, int kt,
                                             int nt, int K) {
  const int n     = nt * 16 + (lane & 15);
  const int kbase = kt * 32 + ((lane < 16) ? 0 : 8);
  v16bf b;
#pragma unroll
  for (int j = 0; j < 8; ++j) {
    const int k  = kbase + ((j < 4) ? (2 * j) : (16 + 2 * (j - 4)));
    b[2 * j]     = W[n * K + k];
    b[2 * j + 1] = W[n * K + k + 1];
  }
  return b;
}

// C/D layout: vgpr r, lane L -> m = (L<16 ? r : r+8), n = nt*16 + (L&15).
__device__ __forceinline__ v8f bias_acc(const float* sB, int lane, int nt) {
  const float bv = sB[nt * 16 + (lane & 15)];
  v8f c;
#pragma unroll
  for (int j = 0; j < 8; ++j) c[j] = bv;
  return c;
}

// ---------------------------------------------------------------------------
// TDM: DMA a 64x64 fp32 matrix (row-major, contiguous) from global into LDS.
// D# descriptor per CDNA5 ISA ch.8; lds_addr = low 32 bits of generic LDS ptr
// (generic-LDS addressing: addr[31:0] is the LDS offset).
// ---------------------------------------------------------------------------
__device__ __forceinline__ void tdm_load_64x64_f32(const float* g, float* lds) {
#if defined(__AMDGCN__) && __has_builtin(__builtin_amdgcn_tensor_load_to_lds)
  const unsigned long long ga = (unsigned long long)(uintptr_t)g;
  const unsigned ldsOff = (unsigned)(uintptr_t)lds;
  v4u g0 = { 1u,                                   // count=1, user mode
             ldsOff,                               // lds_addr
             (unsigned)(ga & 0xffffffffu),         // global_addr[31:0]
             (unsigned)((ga >> 32) & 0x01ffffffu)  // global_addr[56:32]
                 | (2u << 30) };                   // type=2 (image)
  v8i_t g1 = { (int)(2u << 16),   // data_size=2 -> 4-byte elements
               (int)(64u << 16),  // tensor_dim0 = 64 (bits 79:48, low half)
               (int)(64u << 16),  // tensor_dim1 = 64 (bits 111:80, low half)
               (int)(64u << 16),  // tile_dim0   = 64 (bits 127:112)
               64,                // tile_dim1   = 64 (bits 143:128)
               64,                // tensor_dim0_stride = 64 (bits 207:160)
               0, 0 };
  v4i_t z4 = { 0, 0, 0, 0 };
#if __clang_major__ >= 23
  v8i_t z8 = { 0, 0, 0, 0, 0, 0, 0, 0 };
  __builtin_amdgcn_tensor_load_to_lds(g0, g1, z4, z4, z8, 0);
#else
  __builtin_amdgcn_tensor_load_to_lds(g0, g1, z4, z4, 0);
#endif
#if __has_builtin(__builtin_amdgcn_s_wait_tensorcnt)
  __builtin_amdgcn_s_wait_tensorcnt(0);
#else
  asm volatile("s_wait_tensorcnt 0x0" ::: "memory");
#endif
#else
  (void)g; (void)lds;
#endif
}

// ---------------------------------------------------------------------------
// Kernel 1: per-node linear  Y(bf16) = X(f32) @ W^T (+ bias).
// One wave = 16 rows; block = 8 waves = 128 rows.
// ---------------------------------------------------------------------------
__global__ __launch_bounds__(256) void node_linear_kernel(
    const float* __restrict__ X, const float* __restrict__ W,
    const float* __restrict__ bias, __bf16* __restrict__ Y, int N) {
  __shared__ __bf16 sW[64 * 64];
  __shared__ float  sB[64];
  __shared__ __bf16 sA[8][16 * 64];
  const int tid = threadIdx.x, lane = tid & 31, wave = tid >> 5;
  for (int i = tid; i < 64 * 64; i += 256) sW[i] = (__bf16)W[i];
  if (tid < 64) sB[tid] = bias ? bias[tid] : 0.0f;
  __syncthreads();

  const int tiles = (N + 127) / 128;
  for (int t = blockIdx.x; t < tiles; t += gridDim.x) {
    const int rowBase = (t * 8 + wave) * 16;
    __bf16* tile = sA[wave];
#pragma unroll
    for (int i = lane; i < 16 * 16; i += 32) {  // 16 rows x 16 float4
      const int r = i >> 4, c4 = (i & 15) << 2;
      int row = rowBase + r; if (row >= N) row = N - 1;
      const float4 f = *(const float4*)(X + (size_t)row * 64 + c4);
      tile[r * 64 + c4 + 0] = (__bf16)f.x;
      tile[r * 64 + c4 + 1] = (__bf16)f.y;
      tile[r * 64 + c4 + 2] = (__bf16)f.z;
      tile[r * 64 + c4 + 3] = (__bf16)f.w;
    }
    asm volatile("s_wait_dscnt 0x0" ::: "memory");

    v8f acc[4];
#pragma unroll
    for (int nt = 0; nt < 4; ++nt) acc[nt] = bias_acc(sB, lane, nt);
#pragma unroll
    for (int kt = 0; kt < 2; ++kt) {
      const v16bf a = load_a_frag(tile, lane, kt, 64);
#pragma unroll
      for (int nt = 0; nt < 4; ++nt)
        acc[nt] = wmma_bf16(a, load_b_frag(sW, lane, kt, nt, 64), acc[nt]);
    }
    if (rowBase + 16 <= N) {  // full tile: branchless stores
#pragma unroll
      for (int rr = 0; rr < 8; ++rr) {
        const int m = (lane < 16) ? rr : rr + 8;
#pragma unroll
        for (int nt = 0; nt < 4; ++nt)
          Y[(size_t)(rowBase + m) * 64 + nt * 16 + (lane & 15)] =
              (__bf16)acc[nt][rr];
      }
    } else {
#pragma unroll
      for (int rr = 0; rr < 8; ++rr) {
        const int m   = (lane < 16) ? rr : rr + 8;
        const int row = rowBase + m;
        if (row < N) {
#pragma unroll
          for (int nt = 0; nt < 4; ++nt)
            Y[(size_t)row * 64 + nt * 16 + (lane & 15)] = (__bf16)acc[nt][rr];
        }
      }
    }
  }
}

// ---------------------------------------------------------------------------
// Kernel 2: per-edge message.  h = rl[dst] + ll[src] + ef*we ; LN ; ReLU ;
// h @ Wf^T + bf ; atomic scatter-add into agg[dst].
// One wave = 16 edges (2 lanes per edge for the LayerNorm).
// Wf is staged block-wide via the Tensor Data Mover.
// ---------------------------------------------------------------------------
__global__ __launch_bounds__(256) void edge_msg_kernel(
    const __bf16* __restrict__ rl, const __bf16* __restrict__ ll,
    const int* __restrict__ dstI, const int* __restrict__ srcI,
    const float* __restrict__ ef, const float* __restrict__ We,
    const float* __restrict__ g1, const float* __restrict__ b1,
    const float* __restrict__ Wf, const float* __restrict__ bf,
    float* __restrict__ agg, int nE) {
  __shared__ float  sStage[64 * 64];  // fp32 TDM landing zone (16KB)
  __shared__ __bf16 sWf[64 * 64];
  __shared__ float  sWe[64], sG[64], sBn[64], sBf[64];
  __shared__ __bf16 sA[8][16 * 64];
  const int tid = threadIdx.x, lane = tid & 31, wave = tid >> 5;

#if defined(__AMDGCN__) && __has_builtin(__builtin_amdgcn_tensor_load_to_lds)
  if (wave == 0) tdm_load_64x64_f32(Wf, sStage);  // one DMA per block
  __syncthreads();
  for (int i = tid; i < 64 * 64; i += 256) sWf[i] = (__bf16)sStage[i];
#else
  for (int i = tid; i < 64 * 64; i += 256) sWf[i] = (__bf16)Wf[i];
#endif
  if (tid < 64) { sWe[tid] = We[tid]; sG[tid] = g1[tid]; sBn[tid] = b1[tid]; sBf[tid] = bf[tid]; }
  __syncthreads();

  const int tiles = (nE + 127) / 128;
  for (int t = blockIdx.x; t < tiles; t += gridDim.x) {
    const int eBase = (t * 8 + wave) * 16;
    const int r = lane >> 1, half = lane & 1;
    int e = eBase + r; if (e >= nE) e = nE - 1;
    const int   d  = dstI[e];
    const int   s  = srcI[e];
    const float fe = ef[e];
    {  // prefetch next tile's indices (global_prefetch_b8)
      const int en = eBase + (int)gridDim.x * 128;
      if (en < nE) __builtin_prefetch(dstI + en, 0, 0);
    }
    const __bf16* prl = rl + (size_t)d * 64 + half * 32;
    const __bf16* pll = ll + (size_t)s * 64 + half * 32;
    float h[32];
    float sum = 0.0f, sq = 0.0f;
#pragma unroll
    for (int j = 0; j < 32; j += 8) {  // 16B bf16 vector gathers
      const v8bf a8 = *(const v8bf*)(prl + j);
      const v8bf b8 = *(const v8bf*)(pll + j);
#pragma unroll
      for (int u = 0; u < 8; ++u) {
        const int col = half * 32 + j + u;
        const float x = (float)a8[u] + (float)b8[u] + fe * sWe[col];
        h[j + u] = x;
        sum += x;
        sq  += x * x;
      }
    }
    sum += __shfl_xor(sum, 1, 32);
    sq  += __shfl_xor(sq, 1, 32);
    const float mean = sum * (1.0f / 64.0f);
    const float var  = sq * (1.0f / 64.0f) - mean * mean;
    const float rstd = rsqrtf(var + 1e-5f);

    __bf16* tile = sA[wave];
#pragma unroll
    for (int j = 0; j < 32; ++j) {
      const int col = half * 32 + j;
      float x = (h[j] - mean) * rstd * sG[col] + sBn[col];
      x = x > 0.0f ? x : 0.0f;
      tile[r * 64 + col] = (__bf16)x;
    }
    asm volatile("s_wait_dscnt 0x0" ::: "memory");

    v8f acc[4];
#pragma unroll
    for (int nt = 0; nt < 4; ++nt) acc[nt] = bias_acc(sBf, lane, nt);
#pragma unroll
    for (int kt = 0; kt < 2; ++kt) {
      const v16bf a = load_a_frag(tile, lane, kt, 64);
#pragma unroll
      for (int nt = 0; nt < 4; ++nt)
        acc[nt] = wmma_bf16(a, load_b_frag(sWf, lane, kt, nt, 64), acc[nt]);
    }
    // scatter-add segment sum (global_atomic_add_f32; stays in L2)
    if (eBase + 16 <= nE) {  // full tile: dst via intra-wave shuffle
#pragma unroll
      for (int rr = 0; rr < 8; ++rr) {
        const int dd = __shfl(d, 2 * rr + (lane & 16), 32);
#pragma unroll
        for (int nt = 0; nt < 4; ++nt)
          atomicAdd(agg + (size_t)dd * 64 + nt * 16 + (lane & 15), acc[nt][rr]);
      }
    } else {
#pragma unroll
      for (int rr = 0; rr < 8; ++rr) {
        const int m  = (lane < 16) ? rr : rr + 8;
        const int ee = eBase + m;
        if (ee < nE) {
          const int dd = dstI[ee];
#pragma unroll
          for (int nt = 0; nt < 4; ++nt)
            atomicAdd(agg + (size_t)dd * 64 + nt * 16 + (lane & 15), acc[nt][rr]);
        }
      }
    }
  }
}

// ---------------------------------------------------------------------------
// Kernel 3: output MLP.  z = [LN(agg)*g2+b2 , right] (16x128) ;
// o = relu(z @ Wo1^T + bo1) @ Wo2^T + bo2 ; right += o.
// ---------------------------------------------------------------------------
__global__ __launch_bounds__(256) void node_out_kernel(
    const float* __restrict__ agg, float* __restrict__ right,
    const float* __restrict__ g2, const float* __restrict__ b2,
    const float* __restrict__ Wo1, const float* __restrict__ bo1,
    const float* __restrict__ Wo2, const float* __restrict__ bo2, int N) {
  __shared__ __bf16 sW1[64 * 128];
  __shared__ __bf16 sW2[64 * 64];
  __shared__ float  sG[64], sB[64], sBo1[64], sBo2[64];
  __shared__ __bf16 sZ[8][16 * 128];
  __shared__ __bf16 sT[8][16 * 64];
  const int tid = threadIdx.x, lane = tid & 31, wave = tid >> 5;
  for (int i = tid; i < 64 * 128; i += 256) sW1[i] = (__bf16)Wo1[i];
  for (int i = tid; i < 64 * 64;  i += 256) sW2[i] = (__bf16)Wo2[i];
  if (tid < 64) { sG[tid] = g2[tid]; sB[tid] = b2[tid]; sBo1[tid] = bo1[tid]; sBo2[tid] = bo2[tid]; }
  __syncthreads();

  const int tiles = (N + 127) / 128;
  for (int t = blockIdx.x; t < tiles; t += gridDim.x) {
    const int rowBase = (t * 8 + wave) * 16;
    const int r = lane >> 1, half = lane & 1;
    int row = rowBase + r; if (row >= N) row = N - 1;

    const float* pa = agg + (size_t)row * 64 + half * 32;
    float a[32];
    float sum = 0.0f, sq = 0.0f;
#pragma unroll
    for (int j = 0; j < 32; j += 4) {
      const float4 f = *(const float4*)(pa + j);
      a[j] = f.x; a[j + 1] = f.y; a[j + 2] = f.z; a[j + 3] = f.w;
      sum += f.x + f.y + f.z + f.w;
      sq  += f.x * f.x + f.y * f.y + f.z * f.z + f.w * f.w;
    }
    sum += __shfl_xor(sum, 1, 32);
    sq  += __shfl_xor(sq, 1, 32);
    const float mean = sum * (1.0f / 64.0f);
    const float var  = sq * (1.0f / 64.0f) - mean * mean;
    const float rstd = rsqrtf(var + 1e-5f);

    __bf16* zt = sZ[wave];
#pragma unroll
    for (int j = 0; j < 32; ++j) {
      const int col = half * 32 + j;
      zt[r * 128 + col] = (__bf16)((a[j] - mean) * rstd * sG[col] + sB[col]);
    }
    const float* pr = right + (size_t)row * 64 + half * 32;
#pragma unroll
    for (int j = 0; j < 32; j += 4) {
      const float4 f = *(const float4*)(pr + j);
      const int col = 64 + half * 32 + j;
      zt[r * 128 + col + 0] = (__bf16)f.x;
      zt[r * 128 + col + 1] = (__bf16)f.y;
      zt[r * 128 + col + 2] = (__bf16)f.z;
      zt[r * 128 + col + 3] = (__bf16)f.w;
    }
    asm volatile("s_wait_dscnt 0x0" ::: "memory");

    // GEMM 1: (16x128) @ Wo1^T -> 16x64, + bo1, relu
    v8f acc[4];
#pragma unroll
    for (int nt = 0; nt < 4; ++nt) acc[nt] = bias_acc(sBo1, lane, nt);
#pragma unroll
    for (int kt = 0; kt < 4; ++kt) {
      const v16bf av = load_a_frag(zt, lane, kt, 128);
#pragma unroll
      for (int nt = 0; nt < 4; ++nt)
        acc[nt] = wmma_bf16(av, load_b_frag(sW1, lane, kt, nt, 128), acc[nt]);
    }
    __bf16* tt = sT[wave];
#pragma unroll
    for (int rr = 0; rr < 8; ++rr) {
      const int m = (lane < 16) ? rr : rr + 8;
#pragma unroll
      for (int nt = 0; nt < 4; ++nt) {
        float x = acc[nt][rr];
        x = x > 0.0f ? x : 0.0f;
        tt[m * 64 + nt * 16 + (lane & 15)] = (__bf16)x;
      }
    }
    asm volatile("s_wait_dscnt 0x0" ::: "memory");

    // GEMM 2: (16x64) @ Wo2^T -> 16x64, + bo2 ; residual into right
    v8f acc2[4];
#pragma unroll
    for (int nt = 0; nt < 4; ++nt) acc2[nt] = bias_acc(sBo2, lane, nt);
#pragma unroll
    for (int kt = 0; kt < 2; ++kt) {
      const v16bf av = load_a_frag(tt, lane, kt, 64);
#pragma unroll
      for (int nt = 0; nt < 4; ++nt)
        acc2[nt] = wmma_bf16(av, load_b_frag(sW2, lane, kt, nt, 64), acc2[nt]);
    }
    if (rowBase + 16 <= N) {  // full tile: branchless residual update
#pragma unroll
      for (int rr = 0; rr < 8; ++rr) {
        const int m = (lane < 16) ? rr : rr + 8;
#pragma unroll
        for (int nt = 0; nt < 4; ++nt) {
          float* p = right + (size_t)(rowBase + m) * 64 + nt * 16 + (lane & 15);
          *p = *p + acc2[nt][rr];
        }
      }
    } else {
#pragma unroll
      for (int rr = 0; rr < 8; ++rr) {
        const int m  = (lane < 16) ? rr : rr + 8;
        const int ro = rowBase + m;
        if (ro < N) {
#pragma unroll
          for (int nt = 0; nt < 4; ++nt) {
            float* p = right + (size_t)ro * 64 + nt * 16 + (lane & 15);
            *p = *p + acc2[nt][rr];
          }
        }
      }
    }
  }
}

// ---------------------------------------------------------------------------
// Host orchestration
// ---------------------------------------------------------------------------
extern "C" void kernel_launch(void* const* d_in, const int* in_sizes, int n_in,
                              void* d_out, int out_size, void* d_ws, size_t ws_size,
                              hipStream_t stream) {
  (void)in_sizes; (void)n_in; (void)out_size; (void)ws_size;
  const int NC = 50000, NV = 100000, NE = 1600000;

  const float* cfeat = (const float*)d_in[0];
  const float* vfeat = (const float*)d_in[1];
  const int*   eidx  = (const int*)d_in[2];
  const float* ef    = (const float*)d_in[3];
  const float* Wl  = (const float*)d_in[4];
  const float* bl  = (const float*)d_in[5];
  const float* We  = (const float*)d_in[6];
  const float* Wr  = (const float*)d_in[7];
  const float* g1  = (const float*)d_in[8];
  const float* b1  = (const float*)d_in[9];
  const float* Wf  = (const float*)d_in[10];
  const float* bf  = (const float*)d_in[11];
  const float* g2  = (const float*)d_in[12];
  const float* b2  = (const float*)d_in[13];
  const float* Wo1 = (const float*)d_in[14];
  const float* bo1 = (const float*)d_in[15];
  const float* Wo2 = (const float*)d_in[16];
  const float* bo2 = (const float*)d_in[17];

  // d_out holds the evolving features: [c (50000x64) | v (100000x64)]
  float* c = (float*)d_out;
  float* v = c + (size_t)NC * 64;
  // scratch: rl/ll in bf16 (halves edge-gather bytes), agg fp32 for atomics
  __bf16* rl  = (__bf16*)d_ws;
  __bf16* ll  = rl + (size_t)NV * 64;
  float*  agg = (float*)(ll + (size_t)NV * 64);

  hipMemcpyAsync(c, cfeat, (size_t)NC * 64 * sizeof(float),
                 hipMemcpyDeviceToDevice, stream);
  hipMemcpyAsync(v, vfeat, (size_t)NV * 64 * sizeof(float),
                 hipMemcpyDeviceToDevice, stream);

  const int* ci = eidx;        // constraint index per edge
  const int* vi = eidx + NE;   // variable index per edge
  const dim3 blk(256);

  for (int k = 0; k < 4; ++k) {
    const bool vtoc = ((k & 1) == 0);
    const int NR = vtoc ? NC : NV;
    const int NL = vtoc ? NV : NC;
    float* right = vtoc ? c : v;
    float* left  = vtoc ? v : c;
    const int* dst = vtoc ? ci : vi;
    const int* src = vtoc ? vi : ci;

    const int gR = (NR + 127) / 128;
    const int gL = (NL + 127) / 128;

    node_linear_kernel<<<gR, blk, 0, stream>>>(
        right, Wl + (size_t)k * 4096, bl + k * 64, rl, NR);
    node_linear_kernel<<<gL, blk, 0, stream>>>(
        left, Wr + (size_t)k * 4096, (const float*)nullptr, ll, NL);

    hipMemsetAsync(agg, 0, (size_t)NR * 64 * sizeof(float), stream);

    edge_msg_kernel<<<2048, blk, 0, stream>>>(
        rl, ll, dst, src, ef, We + k * 64, g1 + k * 64, b1 + k * 64,
        Wf + (size_t)k * 4096, bf + k * 64, agg, NE);

    node_out_kernel<<<gR, blk, 0, stream>>>(
        agg, right, g2 + k * 64, b2 + k * 64, Wo1 + (size_t)k * 8192,
        bo1 + k * 64, Wo2 + (size_t)k * 4096, bo2 + k * 64, NR);
  }
}
